// TrittentionCube_55448027792002
// MI455X (gfx1250) — matrix operand order repaired and checked by reading, not compiled, not run on hardware
//
#include <hip/hip_runtime.h>

// ---------------------------------------------------------------------------
// TrittentionCube for MI455X (gfx1250, wave32, WMMA 16x16x32 bf16)
//
//   B=16, S=512, HID=2048, H=4, D=512 (S == D)
//   out = merge_heads( ((softmax((q k^T ck^T) * 1/sqrt(D)) v) cv) )
//
// All matmuls run as C = A * B^T on v_wmma_f32_16x16x32_bf16.
// Double-buffered LDS (buffer index is compile-time constant -> ds_store,
// never flat_store); bf16 GEMMs use GLOBAL_LOAD_ASYNC_TO_LDS_B128 (ASYNCcnt);
// fp32 projections stage through registers with f32->bf16 conversion,
// software-pipelined across K-tiles.
// ---------------------------------------------------------------------------

typedef __attribute__((ext_vector_type(16))) __bf16 bf16x16;
typedef __attribute__((ext_vector_type(8)))  __bf16 bf16x8;
typedef __attribute__((ext_vector_type(4)))  __bf16 bf16x4;
typedef __attribute__((ext_vector_type(8)))  float  floatx8;
typedef __attribute__((ext_vector_type(4)))  float  floatx4;

// async-copy builtin operand types: int4 pointers in global (AS1) / LDS (AS3)
typedef __attribute__((ext_vector_type(4))) int intx4;
typedef __attribute__((address_space(1))) intx4 gintx4;
typedef __attribute__((address_space(3))) intx4 lintx4;
// explicit global-address-space element types for epilogue stores
typedef __attribute__((address_space(1))) __bf16 g_bf16;
typedef __attribute__((address_space(1))) float  g_float;
typedef __attribute__((address_space(1))) const float gc_float;

#define LDSTRIDE 40   // 32 bf16 payload + 8 pad (80B rows, 16B aligned)

#if defined(__has_builtin)
#if __has_builtin(__builtin_amdgcn_global_load_async_to_lds_b128)
#define HAVE_ASYNC_COPY 1
#else
#define HAVE_ASYNC_COPY 0
#endif
#if __has_builtin(__builtin_amdgcn_s_wait_asynccnt)
#define WAIT_ASYNC() __builtin_amdgcn_s_wait_asynccnt(0)
#else
#define WAIT_ASYNC() asm volatile("s_wait_asynccnt 0x0" ::: "memory")
#endif
#else
#define HAVE_ASYNC_COPY 0
#define WAIT_ASYNC() asm volatile("s_wait_asynccnt 0x0" ::: "memory")
#endif

// ---- async 32B global -> LDS copy (two b128, ASYNCcnt-tracked) -------------
__device__ __forceinline__ void async_cp32(const __bf16* g, __bf16* l) {
#if HAVE_ASYNC_COPY
    __builtin_amdgcn_global_load_async_to_lds_b128((gintx4*)g, (lintx4*)l, 0, 0);
    __builtin_amdgcn_global_load_async_to_lds_b128((gintx4*)g, (lintx4*)l, 16, 0);
#else
    (void)g; (void)l;
#endif
}

// ---- register-staged 16-element sliver (for software pipelining) -----------
template <typename TIN> struct Sliver;
template <> struct Sliver<float>  { floatx4 d[4]; };
template <> struct Sliver<__bf16> { bf16x8  d[2]; };

template <typename TIN>
__device__ __forceinline__ Sliver<TIN> fetch16(const TIN* __restrict__ src) {
    Sliver<TIN> s;
    if constexpr (sizeof(TIN) == 4) {
        const floatx4* p = (const floatx4*)src;
        s.d[0] = p[0]; s.d[1] = p[1]; s.d[2] = p[2]; s.d[3] = p[3];
    } else {
        const bf16x8* p = (const bf16x8*)src;
        s.d[0] = p[0]; s.d[1] = p[1];
    }
    return s;
}

template <typename TIN>
__device__ __forceinline__ void commit16(const Sliver<TIN>& s, __bf16* dst) {
    if constexpr (sizeof(TIN) == 4) {
        bf16x8 lo, hi;
#pragma unroll
        for (int i = 0; i < 4; ++i) { lo[i] = (__bf16)s.d[0][i]; lo[4 + i] = (__bf16)s.d[1][i]; }
#pragma unroll
        for (int i = 0; i < 4; ++i) { hi[i] = (__bf16)s.d[2][i]; hi[4 + i] = (__bf16)s.d[3][i]; }
        *(bf16x8*)dst       = lo;
        *(bf16x8*)(dst + 8) = hi;
    } else {
        *(bf16x8*)dst       = s.d[0];
        *(bf16x8*)(dst + 8) = s.d[1];
    }
}

// ---- fragment loads from LDS (wave32 WMMA 16-bit operand layouts) ----------
// A (16x32, MxK): lane m=lane&15, half=lane>>4;
//   elems 0..7  -> K = half*8 .. half*8+7        (ds_read_b128)
//   elems 8..15 -> K = 16+half*8 .. 16+half*8+7  (ds_read_b128)
__device__ __forceinline__ bf16x16 load_frag_A(const __bf16* lds, int rowBase, int lane) {
    const int m = lane & 15, half = lane >> 4;
    const __bf16* p = lds + (rowBase + m) * LDSTRIDE;
    bf16x8 lo = *(const bf16x8*)(p + half * 8);
    bf16x8 hi = *(const bf16x8*)(p + 16 + half * 8);
    return __builtin_shufflevector(lo, hi, 0, 1, 2, 3, 4, 5, 6, 7,
                                   8, 9, 10, 11, 12, 13, 14, 15);
}

// B (32x16, KxN) sourced from row-major [N x K] storage (B^T rows):
// lane n=lane&15, half=lane>>4; elems 0..7 -> K = half*8+e, elems 8..15 -> K = 16+half*8+(e-8)
__device__ __forceinline__ bf16x16 load_frag_B(const __bf16* lds, int rowBase, int lane) {
    const int n = lane & 15, half = lane >> 4;
    const __bf16* p = lds + (rowBase + n) * LDSTRIDE;
    bf16x8 lo = *(const bf16x8*)(p + half * 8);
    bf16x8 hi = *(const bf16x8*)(p + 16 + half * 8);
    return __builtin_shufflevector(lo, hi, 0, 1, 2, 3, 4, 5, 6, 7,
                                   8, 9, 10, 11, 12, 13, 14, 15);
}

// ---------------------------------------------------------------------------
// Batched C = A * B^T.  A: [M x K] row-major per batch, B: [N x K] row-major
// per batch.  Block tile 128x128, 8 waves (2x4), wave tile 64x32, K-step 32,
// double-buffered LDS (K-loop unrolled by 2 so buffer indices are constants),
// one barrier per K-tile.  K must be a multiple of 64.
//
// OUT_MODE:
//   0: bf16, per-head layout        idx = ((b*4+h)*512 + s)*512 + d   (proj)
//   1: bf16, per-head TRANSPOSED    idx = ((b*4+h)*512 + d)*512 + s   (v, cv)
//   2: bf16, batched row-major      idx = z*M*N + row*N + col
//   3: f32,  batched row-major      (s2, feeds softmax)
//   4: f32,  final merge-heads      idx = ((z>>2)*512+row)*2048 + (z&3)*512+col
// ---------------------------------------------------------------------------
template <typename TIN, int OUT_MODE>
__global__ __launch_bounds__(256)
void wmma_gemm_abT(const TIN* __restrict__ A, const TIN* __restrict__ B,
                   void* __restrict__ Cout, const float* __restrict__ bias,
                   float scale, int M, int N, int K,
                   size_t strideA, size_t strideB) {
    constexpr bool ASYNC = (sizeof(TIN) == 2) && (HAVE_ASYNC_COPY != 0);

    __shared__ __bf16 As[2][128 * LDSTRIDE];
    __shared__ __bf16 Bs[2][128 * LDSTRIDE];

    const int t    = threadIdx.x;
    const int lane = t & 31;
    const int wid  = t >> 5;
    const int waveM = wid >> 2;   // 0..1  (64 rows each)
    const int waveN = wid & 3;    // 0..3  (32 cols each)
    const int z        = blockIdx.z;
    const int tileRow0 = blockIdx.y * 128;
    const int tileCol0 = blockIdx.x * 128;

    // staging coords: 256 threads cover 128 rows x 2 halves of 16 elements
    const int sr = t >> 1;
    const int sc = (t & 1) << 4;
    const int stageOff = sr * LDSTRIDE + sc;

    const TIN* Agp = A + (size_t)z * strideA + (size_t)(tileRow0 + sr) * K + sc;
    const TIN* Bgp = B + (size_t)z * strideB + (size_t)(tileCol0 + sr) * K + sc;

    // ---- prologue: fill buffer 0 ------------------------------------------
    if constexpr (ASYNC) {
        async_cp32((const __bf16*)Agp, &As[0][stageOff]);
        async_cp32((const __bf16*)Bgp, &Bs[0][stageOff]);
        WAIT_ASYNC();
    } else {
        commit16<TIN>(fetch16<TIN>(Agp), &As[0][stageOff]);
        commit16<TIN>(fetch16<TIN>(Bgp), &Bs[0][stageOff]);
    }
    __syncthreads();

    floatx8 acc[4][2] = {};

    // one pipeline step: compute on buffer cb (constant), stage kc+32 into cb^1
    auto step = [&](const int cb, const int kc) {
        const int kn = kc + 32;
        const bool has_next = kn < K;
        Sliver<TIN> sa, sb;

        if (has_next) {
            if constexpr (ASYNC) {
                async_cp32((const __bf16*)(Agp + kn), &As[cb ^ 1][stageOff]);
                async_cp32((const __bf16*)(Bgp + kn), &Bs[cb ^ 1][stageOff]);
            } else {
                sa = fetch16<TIN>(Agp + kn);
                sb = fetch16<TIN>(Bgp + kn);
                if (kn + 32 < K) {  // gfx1250 global_prefetch_b8, 2 tiles ahead
                    __builtin_prefetch(Agp + kn + 32, 0, 0);
                    __builtin_prefetch(Bgp + kn + 32, 0, 0);
                }
            }
        }

        bf16x16 bfr0 = load_frag_B(&Bs[cb][0], waveN * 32, lane);
        bf16x16 bfr1 = load_frag_B(&Bs[cb][0], waveN * 32 + 16, lane);
#pragma unroll
        for (int mi = 0; mi < 4; ++mi) {
            bf16x16 afr = load_frag_A(&As[cb][0], waveM * 64 + mi * 16, lane);
            acc[mi][0] = __builtin_amdgcn_wmma_f32_16x16x32_bf16(
                false, afr, false, bfr0, (short)0, acc[mi][0], false, false);
            acc[mi][1] = __builtin_amdgcn_wmma_f32_16x16x32_bf16(
                false, afr, false, bfr1, (short)0, acc[mi][1], false, false);
        }

        if (has_next) {
            if constexpr (ASYNC) {
                WAIT_ASYNC();
            } else {
                commit16<TIN>(sa, &As[cb ^ 1][stageOff]);
                commit16<TIN>(sb, &Bs[cb ^ 1][stageOff]);
            }
        }
        __syncthreads();
    };

    for (int k0 = 0; k0 < K; k0 += 64) {  // K % 64 == 0 for all call sites
        step(0, k0);
        step(1, k0 + 32);
    }

    // ---- epilogue: scale (+bias), convert, scatter per OUT_MODE -----------
    const int nlane = lane & 15;
    const int half  = lane >> 4;
#pragma unroll
    for (int mi = 0; mi < 4; ++mi) {
#pragma unroll
        for (int ni = 0; ni < 2; ++ni) {
            const int colT = tileCol0 + waveN * 32 + ni * 16 + nlane;
            const float bv = bias ? ((gc_float*)bias)[colT] : 0.0f;
#pragma unroll
            for (int i = 0; i < 8; ++i) {
                const int rowT = tileRow0 + waveM * 64 + mi * 16 + half * 8 + i;
                const float v = acc[mi][ni][i] * scale + bv;
                if constexpr (OUT_MODE == 0) {
                    size_t idx = ((size_t)(((rowT >> 9) * 4 + (colT >> 9)) * 512 +
                                           (rowT & 511))) * 512 + (colT & 511);
                    ((g_bf16*)Cout)[idx] = (__bf16)v;
                } else if constexpr (OUT_MODE == 1) {
                    size_t idx = ((size_t)(((rowT >> 9) * 4 + (colT >> 9)) * 512 +
                                           (colT & 511))) * 512 + (rowT & 511);
                    ((g_bf16*)Cout)[idx] = (__bf16)v;
                } else if constexpr (OUT_MODE == 2) {
                    size_t idx = (size_t)z * ((size_t)M * (size_t)N) +
                                 (size_t)rowT * (size_t)N + colT;
                    ((g_bf16*)Cout)[idx] = (__bf16)v;
                } else if constexpr (OUT_MODE == 3) {
                    size_t idx = (size_t)z * ((size_t)M * (size_t)N) +
                                 (size_t)rowT * (size_t)N + colT;
                    ((g_float*)Cout)[idx] = v;
                } else {  // OUT_MODE == 4: merge heads into [B,S,HID] f32
                    size_t idx = (size_t)((z >> 2) * 512 + rowT) * 2048 +
                                 (size_t)((z & 3) * 512 + colT);
                    ((g_float*)Cout)[idx] = v;
                }
            }
        }
    }
}

// ---------------------------------------------------------------------------
// Row softmax over 512 f32 elements -> bf16.  One wave32 per row, 8 rows/block.
// ---------------------------------------------------------------------------
__global__ __launch_bounds__(256)
void softmax512_bf16(const float* __restrict__ in, __bf16* __restrict__ out,
                     int nrows) {
    const int lane = threadIdx.x & 31;
    const int wid  = threadIdx.x >> 5;
    const int row  = blockIdx.x * 8 + wid;
    if (row >= nrows) return;

    const floatx4* in4 = (const floatx4*)(in + (size_t)row * 512);
    floatx4 v[4];
#pragma unroll
    for (int j = 0; j < 4; ++j) v[j] = in4[j * 32 + lane];

    float m = -3.4e38f;
#pragma unroll
    for (int j = 0; j < 4; ++j)
#pragma unroll
        for (int e = 0; e < 4; ++e) m = fmaxf(m, v[j][e]);
#pragma unroll
    for (int off = 16; off > 0; off >>= 1) m = fmaxf(m, __shfl_xor(m, off, 32));

    float s = 0.0f;
#pragma unroll
    for (int j = 0; j < 4; ++j)
#pragma unroll
        for (int e = 0; e < 4; ++e) {
            float ex = __expf(v[j][e] - m);
            v[j][e] = ex;
            s += ex;
        }
#pragma unroll
    for (int off = 16; off > 0; off >>= 1) s += __shfl_xor(s, off, 32);
    const float inv = 1.0f / s;

    bf16x4* out4 = (bf16x4*)(out + (size_t)row * 512);
#pragma unroll
    for (int j = 0; j < 4; ++j) {
        bf16x4 o;
#pragma unroll
        for (int e = 0; e < 4; ++e) o[e] = (__bf16)(v[j][e] * inv);
        out4[j * 32 + lane] = o;
    }
}

// ---------------------------------------------------------------------------
extern "C" void kernel_launch(void* const* d_in, const int* in_sizes, int n_in,
                              void* d_out, int out_size, void* d_ws,
                              size_t ws_size, hipStream_t stream) {
    (void)in_sizes; (void)n_in; (void)out_size; (void)ws_size;

    const float* hs  = (const float*)d_in[0];
    const float* Wq  = (const float*)d_in[1];  const float* bq  = (const float*)d_in[2];
    const float* Wk  = (const float*)d_in[3];  const float* bk  = (const float*)d_in[4];
    const float* Wv  = (const float*)d_in[5];  const float* bv  = (const float*)d_in[6];
    const float* Wck = (const float*)d_in[7];  const float* bck = (const float*)d_in[8];
    const float* Wcv = (const float*)d_in[9];  const float* bcv = (const float*)d_in[10];

    const size_t PROJ = (size_t)16 * 4 * 512 * 512;  // 16,777,216 elems / head buf
    __bf16* Qh   = (__bf16*)d_ws;
    __bf16* Kh   = Qh + PROJ;
    __bf16* CKh  = Kh + PROJ;
    __bf16* VhT  = CKh + PROJ;      // stored transposed: [b,h,d,s]
    __bf16* CVhT = VhT + PROJ;      // stored transposed: [b,h,e,m]
    __bf16* S1   = CVhT + PROJ;
    // reuse (stream-ordered, producers consumed before overwrite):
    float*  S2f  = (float*)d_ws;    // overlays Qh+Kh (67.1 MB)
    __bf16* P    = CKh;             // overlays CKh after s2
    __bf16* CTX1 = S1;              // overlays S1 after softmax/ctx1

    const float SCALE = 0.044194173824159216f;  // 1/sqrt(512)
    const dim3 blk(256);
    const dim3 gProj(2048 / 128, 8192 / 128, 1);   // N tiles, M tiles
    const dim3 gAttn(512 / 128, 512 / 128, 64);    // batched per (b,h)
    const size_t SB = (size_t)512 * 512;           // batch stride, attention

    // --- projections: [8192x2048] * W^T (+bias), fp32 -> bf16 heads ---------
    wmma_gemm_abT<float, 0><<<gProj, blk, 0, stream>>>(hs, Wq,  Qh,   bq,  1.0f, 8192, 2048, 2048, 0, 0);
    wmma_gemm_abT<float, 0><<<gProj, blk, 0, stream>>>(hs, Wk,  Kh,   bk,  1.0f, 8192, 2048, 2048, 0, 0);
    wmma_gemm_abT<float, 0><<<gProj, blk, 0, stream>>>(hs, Wck, CKh,  bck, 1.0f, 8192, 2048, 2048, 0, 0);
    wmma_gemm_abT<float, 1><<<gProj, blk, 0, stream>>>(hs, Wv,  VhT,  bv,  1.0f, 8192, 2048, 2048, 0, 0);
    wmma_gemm_abT<float, 1><<<gProj, blk, 0, stream>>>(hs, Wcv, CVhT, bcv, 1.0f, 8192, 2048, 2048, 0, 0);

    // --- s1 = q k^T (bf16) --------------------------------------------------
    wmma_gemm_abT<__bf16, 2><<<gAttn, blk, 0, stream>>>(Qh, Kh, S1, nullptr, 1.0f, 512, 512, 512, SB, SB);
    // --- s2 = (s1 ck^T) * SCALE (f32, feeds softmax) ------------------------
    wmma_gemm_abT<__bf16, 3><<<gAttn, blk, 0, stream>>>(S1, CKh, S2f, nullptr, SCALE, 512, 512, 512, SB, SB);
    // --- probs = softmax(s2) (bf16) ----------------------------------------
    softmax512_bf16<<<dim3(64 * 512 / 8), blk, 0, stream>>>(S2f, P, 64 * 512);
    // --- ctx1 = probs @ v  (via v^T stored row-major) -----------------------
    wmma_gemm_abT<__bf16, 2><<<gAttn, blk, 0, stream>>>(P, VhT, CTX1, nullptr, 1.0f, 512, 512, 512, SB, SB);
    // --- ctx2 = ctx1 @ cv -> merge heads, f32 to d_out ----------------------
    wmma_gemm_abT<__bf16, 4><<<gAttn, blk, 0, stream>>>(CTX1, CVhT, (float*)d_out, nullptr, 1.0f, 512, 512, 512, SB, SB);
}
